// AllegroLayer_26534307954737
// MI455X (gfx1250) — matrix operand order repaired and checked by reading, not compiled, not run on hardware
//
#include <hip/hip_runtime.h>
#include <math.h>

// ---------------------------------------------------------------------------
// Types for gfx1250 WMMA (wave32): v_wmma_f32_16x16x32_bf16
// ---------------------------------------------------------------------------
typedef __bf16 bf16_t;
typedef bf16_t v16bf __attribute__((ext_vector_type(16)));
typedef float  v8f   __attribute__((ext_vector_type(8)));

union FragU { v16bf v; unsigned short s[16]; };

__device__ __forceinline__ unsigned short bfbits(float f) {
  unsigned u = __builtin_bit_cast(unsigned, f);
  u += 0x7FFFu + ((u >> 16) & 1u);            // round-to-nearest-even
  return (unsigned short)(u >> 16);
}

__device__ __forceinline__ v8f zero8() {
  v8f a;
  #pragma unroll
  for (int i = 0; i < 8; ++i) a[i] = 0.f;
  return a;
}

__device__ __forceinline__ v8f wmma_bf(v16bf a, v16bf b, v8f c) {
  return __builtin_amdgcn_wmma_f32_16x16x32_bf16(false, a, false, b, (short)0, c,
                                                 false, false);
}

// A-fragment (16x32 bf16) from row-major f32: lane L holds M=L%16,
// K = khalf + 0..7 in s[0..7], K = 16+khalf+0..7 in s[8..15], khalf = L<16?0:8
__device__ __forceinline__ v16bf ldA_f32(const float* __restrict__ base, int ld,
                                         int kbase, int lane) {
  FragU f;
  const float* row = base + (lane & 15) * ld + kbase + ((lane & 16) ? 8 : 0);
  #pragma unroll
  for (int j = 0; j < 8; ++j) f.s[j] = bfbits(row[j]);
  #pragma unroll
  for (int j = 0; j < 8; ++j) f.s[8 + j] = bfbits(row[16 + j]);
  return f.v;
}

// same but source already bf16 (LDS staging), row-major
__device__ __forceinline__ v16bf ldA_bf(const unsigned short* base, int ld,
                                        int kbase, int lane) {
  FragU f;
  const unsigned short* row = base + (lane & 15) * ld + kbase + ((lane & 16) ? 8 : 0);
  #pragma unroll
  for (int j = 0; j < 8; ++j) f.s[j] = row[j];
  #pragma unroll
  for (int j = 0; j < 8; ++j) f.s[8 + j] = row[16 + j];
  return f.v;
}

// B-fragment pre-packed: 32 lanes x 16 bf16 contiguous per fragment
__device__ __forceinline__ v16bf ldB(const unsigned short* __restrict__ packed,
                                     int frag, int lane) {
  FragU f;
  const unsigned short* p = packed + ((size_t)frag * 32 + lane) * 16;
  #pragma unroll
  for (int j = 0; j < 16; ++j) f.s[j] = p[j];
  return f.v;
}

// ---------------------------------------------------------------------------
// Constants layout in d_ws (floats): CG tables computed on device
// ---------------------------------------------------------------------------
#define OFF_REC0 0     // w3j(1,1,2)  45
#define OFF_REC1 45    // w3j(2,1,3) 105
#define OFF_S0   150
#define OFF_S1   151
#define OFF_SILU 152
#define OFF_P000 160
#define OFF_P110 161
#define OFF_P220 170
#define OFF_P011 195
#define OFF_P101 204
#define OFF_P121 213
#define OFF_P211 258
#define OFF_P321 303
#define OFF_P022 408
#define OFF_P112 433
#define OFF_P202 478
#define OFF_P222 503
#define OFF_P312 628
#define NCONSTS  768

// ---------------------------------------------------------------------------
// Device-side Wigner 3j (e3nn real basis)
// ---------------------------------------------------------------------------
struct cplx { double re, im; };
__device__ __forceinline__ cplx cmul(cplx a, cplx b) {
  cplx r; r.re = a.re * b.re - a.im * b.im; r.im = a.re * b.im + a.im * b.re; return r;
}
__device__ __forceinline__ int imax3(int a, int b, int c) {
  int m = a > b ? a : b; return m > c ? m : c;
}
__device__ __forceinline__ int imin3(int a, int b, int c) {
  int m = a < b ? a : b; return m < c ? m : c;
}
__device__ double dfact(int n) { double r = 1.0; for (int i = 2; i <= n; ++i) r *= (double)i; return r; }

__device__ double su2cg(int j1, int m1, int j2, int m2, int j3, int m3) {
  if (m3 != m1 + m2) return 0.0;
  int vmin = imax3(-j1 + j2 + m3, -j1 + m1, 0);
  int vmax = imin3(j2 + j3 + m1, j3 - j1 + j2, j3 + m3);
  double c = sqrt((2.0 * j3 + 1.0) * dfact(j3 + j1 - j2) * dfact(j3 - j1 + j2) *
                  dfact(j1 + j2 - j3) / dfact(j1 + j2 + j3 + 1) *
                  dfact(j3 + m3) * dfact(j3 - m3) /
                  (dfact(j1 + m1) * dfact(j1 - m1) * dfact(j2 + m2) * dfact(j2 - m2)));
  double s = 0.0;
  for (int v = vmin; v <= vmax; ++v) {
    double sg = ((v + j2 + m2) & 1) ? -1.0 : 1.0;
    s += sg / dfact(v) * dfact(j2 + j3 + m1 - v) * dfact(j1 - m1 + v) /
         dfact(j3 - j1 + j2 - v) / dfact(j3 + m3 - v) / dfact(v + j1 - j2 - m3);
  }
  return c * s;
}

__device__ void qmat(int l, cplx* q) {     // (2l+1)^2, real->complex SH basis
  int n = 2 * l + 1;
  for (int i = 0; i < n * n; ++i) { q[i].re = 0; q[i].im = 0; }
  double is2 = sqrt(0.5);
  for (int m = -l; m < 0; ++m) {
    q[(l + m) * n + (l - m)].re = is2;
    q[(l + m) * n + (l + m)].im = -is2;
  }
  q[l * n + l].re = 1.0;
  for (int m = 1; m <= l; ++m) {
    double sg = (m & 1) ? -1.0 : 1.0;
    q[(l + m) * n + (l + m)].re = sg * is2;
    q[(l + m) * n + (l - m)].im = sg * is2;
  }
  cplx ph; ph.re = 1; ph.im = 0; cplx mi; mi.re = 0; mi.im = -1;
  for (int t = 0; t < l; ++t) ph = cmul(ph, mi);     // (-i)^l
  for (int i = 0; i < n * n; ++i) q[i] = cmul(q[i], ph);
}

__device__ void w3j_dev(int l1, int l2, int l3, float* out) {
  int n1 = 2 * l1 + 1, n2 = 2 * l2 + 1, n3 = 2 * l3 + 1;
  cplx q1[49], q2[49], q3[49];
  qmat(l1, q1); qmat(l2, q2); qmat(l3, q3);
  double C[343];
  for (int i = 0; i < n1 * n2 * n3; ++i) C[i] = 0.0;
  for (int m1 = -l1; m1 <= l1; ++m1)
    for (int m2 = -l2; m2 <= l2; ++m2) {
      int m3 = m1 + m2;
      if (m3 >= -l3 && m3 <= l3)
        C[((l1 + m1) * n2 + (l2 + m2)) * n3 + (l3 + m3)] = su2cg(l1, m1, l2, m2, l3, m3);
    }
  double R[343]; double nrm = 0.0;
  for (int j = 0; j < n1; ++j)
    for (int l = 0; l < n2; ++l)
      for (int nn = 0; nn < n3; ++nn) {
        double acc = 0.0;
        for (int i = 0; i < n1; ++i)
          for (int k = 0; k < n2; ++k)
            for (int m = 0; m < n3; ++m) {
              double cc = C[(i * n2 + k) * n3 + m];
              if (cc == 0.0) continue;
              cplx t = cmul(q1[i * n1 + j], q2[k * n2 + l]);
              cplx q3c; q3c.re = q3[m * n3 + nn].re; q3c.im = -q3[m * n3 + nn].im;
              t = cmul(t, q3c);
              acc += cc * t.re;
            }
        R[(j * n2 + l) * n3 + nn] = acc; nrm += acc * acc;
      }
  double s = 1.0 / sqrt(nrm);
  for (int i = 0; i < n1 * n2 * n3; ++i) out[i] = (float)(R[i] * s);
}

__global__ void build_consts(float* c) {
  if (threadIdx.x != 0 || blockIdx.x != 0) return;
  w3j_dev(1, 1, 2, c + OFF_REC0);
  w3j_dev(2, 1, 3, c + OFF_REC1);
  // SH recursion scales (reference _sh_scales)
  double n[3] = {0.3, 0.5, 0.81};
  double nl = sqrt(n[0] * n[0] + n[1] * n[1] + n[2] * n[2]);
  double ny[3] = {n[1] / nl, n[2] / nl, n[0] / nl};
  double Y[7]; for (int i = 0; i < 3; ++i) Y[i] = sqrt(3.0) * ny[i];
  double A[7];
  for (int k = 0; k < 5; ++k) A[k] = 0;
  for (int i = 0; i < 3; ++i)
    for (int j = 0; j < 3; ++j) {
      double p = Y[i] * ny[j];
      for (int k = 0; k < 5; ++k) A[k] += p * (double)c[OFF_REC0 + (i * 3 + j) * 5 + k];
    }
  double na = 0; for (int k = 0; k < 5; ++k) na += A[k] * A[k];
  double s0 = sqrt(5.0) / sqrt(na);
  for (int k = 0; k < 5; ++k) Y[k] = A[k] * s0;
  for (int k = 0; k < 7; ++k) A[k] = 0;
  for (int i = 0; i < 5; ++i)
    for (int j = 0; j < 3; ++j) {
      double p = Y[i] * ny[j];
      for (int k = 0; k < 7; ++k) A[k] += p * (double)c[OFF_REC1 + (i * 3 + j) * 7 + k];
    }
  na = 0; for (int k = 0; k < 7; ++k) na += A[k] * A[k];
  double s1 = sqrt(7.0) / sqrt(na);
  c[OFF_S0] = (float)s0; c[OFF_S1] = (float)s1;
  // normalized-silu constant: 1/sqrt(E[silu(z)^2]), trapezoid on [-12,12]
  double sum = 0.0; const int N = 24000; double dz = 24.0 / N;
  for (int i = 0; i <= N; ++i) {
    double z = -12.0 + dz * i;
    double phi = exp(-0.5 * z * z) * 0.3989422804014327;
    double sl = z / (1.0 + exp(-z));
    double f = sl * sl * phi;
    sum += (i == 0 || i == N) ? 0.5 * f : f;
  }
  sum *= dz;
  c[OFF_SILU] = (float)(1.0 / sqrt(sum));
  // tensor-product path CG tables
  w3j_dev(0, 0, 0, c + OFF_P000);
  w3j_dev(1, 1, 0, c + OFF_P110);
  w3j_dev(2, 2, 0, c + OFF_P220);
  w3j_dev(0, 1, 1, c + OFF_P011);
  w3j_dev(1, 0, 1, c + OFF_P101);
  w3j_dev(1, 2, 1, c + OFF_P121);
  w3j_dev(2, 1, 1, c + OFF_P211);
  w3j_dev(3, 2, 1, c + OFF_P321);
  w3j_dev(0, 2, 2, c + OFF_P022);
  w3j_dev(1, 1, 2, c + OFF_P112);
  w3j_dev(2, 0, 2, c + OFF_P202);
  w3j_dev(2, 2, 2, c + OFF_P222);
  w3j_dev(3, 1, 2, c + OFF_P312);
}

// ---------------------------------------------------------------------------
// Pack weight (K x 64, row-major f32) into B-fragment bf16 layout.
// frag = kc*4 + nt ; per fragment: lane L -> N = nt*16 + (L&15),
// K = kc*32 + (L<16?0:8) + j for j<8, K = kc*32 + 16 + (L<16?0:8) + (j-8) for j>=8
// ---------------------------------------------------------------------------
__global__ void pack_b(const float* __restrict__ W, unsigned short* __restrict__ dst,
                       int K) {
  int t = blockIdx.x * blockDim.x + threadIdx.x;
  int wave = t >> 5, lane = t & 31;
  int nf = (K >> 5) * 4;
  if (wave >= nf) return;
  int kc = wave >> 2, nt = wave & 3;
  int kb = kc * 32 + ((lane < 16) ? 0 : 8);
  int n = nt * 16 + (lane & 15);
  unsigned short* o = dst + ((size_t)wave * 32 + lane) * 16;
  #pragma unroll
  for (int j = 0; j < 16; ++j) o[j] = bfbits(W[(size_t)(kb + j + (j >= 8 ? 8 : 0)) * 64 + n]);
}

__global__ void zero_f32(float* p, int n) {
  int i = blockIdx.x * blockDim.x + threadIdx.x;
  int st = gridDim.x * blockDim.x;
  for (; i < n; i += st) p[i] = 0.f;
}

// ---------------------------------------------------------------------------
// Templated dense CG contraction:  out[k] = scale * sum_ij wy[l1^2+i]*vc[l2^2+j]*cg[i,j,k]
// ---------------------------------------------------------------------------
template <int L1, int L2, int L3>
__device__ __forceinline__ void tpath(const float* wy, const float* vc,
                                      const float* cg, float scale, float* out) {
  constexpr int N1 = 2 * L1 + 1, N2 = 2 * L2 + 1, N3 = 2 * L3 + 1;
  const float* a = wy + L1 * L1;
  const float* b = vc + L2 * L2;
  float o[N3];
  #pragma unroll
  for (int k = 0; k < N3; ++k) o[k] = 0.f;
  #pragma unroll
  for (int i = 0; i < N1; ++i) {
    #pragma unroll
    for (int j = 0; j < N2; ++j) {
      float p = a[i] * b[j];
      #pragma unroll
      for (int k = 0; k < N3; ++k) o[k] += p * cg[(i * N2 + j) * N3 + k];
    }
  }
  #pragma unroll
  for (int k = 0; k < N3; ++k) out[k] = o[k] * scale;
}

// ---------------------------------------------------------------------------
// Kernel 1: per 16-edge tile: w = x@W0/sqrt(128) via WMMA, spherical harmonics,
// scatter w (x) Y into node_sum with f32 atomics (L2-resident, 8MB).
// ---------------------------------------------------------------------------
__global__ __launch_bounds__(256) void edge_scatter(
    const float* __restrict__ vec, const float* __restrict__ x,
    const int* __restrict__ senders, const unsigned short* __restrict__ pW0,
    const float* __restrict__ consts, float* __restrict__ node_sum) {
  __shared__ float sY[8][16][16];
  __shared__ int sS[8][16];
  int tid = threadIdx.x, wave = tid >> 5, lane = tid & 31;
  int e0 = (blockIdx.x * 8 + wave) * 16;

  if (lane < 16) {
    int e = e0 + lane;
    sS[wave][lane] = senders[e];
    float vx = vec[e * 3 + 0], vy = vec[e * 3 + 1], vz = vec[e * 3 + 2];
    float inv = rsqrtf(vx * vx + vy * vy + vz * vz);
    float nb[3] = {vy * inv, vz * inv, vx * inv};   // e3nn y,z,x order
    float y1[3], y2[5], y3[7];
    #pragma unroll
    for (int i = 0; i < 3; ++i) y1[i] = 1.7320508075688772f * nb[i];
    #pragma unroll
    for (int k = 0; k < 5; ++k) y2[k] = 0.f;
    #pragma unroll
    for (int i = 0; i < 3; ++i)
      for (int j = 0; j < 3; ++j) {
        float p = y1[i] * nb[j];
        #pragma unroll
        for (int k = 0; k < 5; ++k) y2[k] += p * consts[OFF_REC0 + (i * 3 + j) * 5 + k];
      }
    float sc0 = consts[OFF_S0], sc1 = consts[OFF_S1];
    #pragma unroll
    for (int k = 0; k < 5; ++k) y2[k] *= sc0;
    #pragma unroll
    for (int k = 0; k < 7; ++k) y3[k] = 0.f;
    #pragma unroll
    for (int i = 0; i < 5; ++i)
      for (int j = 0; j < 3; ++j) {
        float p = y2[i] * nb[j];
        #pragma unroll
        for (int k = 0; k < 7; ++k) y3[k] += p * consts[OFF_REC1 + (i * 3 + j) * 7 + k];
      }
    float* Y = sY[wave][lane];
    Y[0] = 1.f;
    #pragma unroll
    for (int i = 0; i < 3; ++i) Y[1 + i] = y1[i];
    #pragma unroll
    for (int k = 0; k < 5; ++k) Y[4 + k] = y2[k];
    #pragma unroll
    for (int k = 0; k < 7; ++k) Y[9 + k] = y3[k] * sc1;
  }

  // w tile (16 edges x 64) via 4x4 WMMA chain
  v8f acc[4];
  #pragma unroll
  for (int nt = 0; nt < 4; ++nt) {
    acc[nt] = zero8();
    #pragma unroll
    for (int kc = 0; kc < 4; ++kc) {
      v16bf a = ldA_f32(x + (size_t)e0 * 128, 128, kc * 32, lane);
      v16bf b = ldB(pW0, kc * 4 + nt, lane);
      acc[nt] = wmma_bf(a, b, acc[nt]);
    }
  }
  __syncthreads();

  const float inv128 = 0.08838834764831845f;  // 1/sqrt(128)
  int hi = (lane & 16) ? 8 : 0;
  #pragma unroll
  for (int nt = 0; nt < 4; ++nt) {
    int u = nt * 16 + (lane & 15);
    #pragma unroll
    for (int r = 0; r < 8; ++r) {
      int el = r + hi;
      float wv = acc[nt][r] * inv128;
      float* base = node_sum + ((size_t)sS[wave][el] * 64 + u) * 16;
      const float* Ye = sY[wave][el];
      #pragma unroll
      for (int i = 0; i < 16; ++i) atomicAdd(base + i, wv * Ye[i]);
    }
  }
}

// ---------------------------------------------------------------------------
// Kernel 2: per 16-edge tile: gather node sums, CG tensor products -> bf16 LDS
// staging, then MLP chain (wave 0) + 32 equivariant WMMA chains (waves 1-3).
// ---------------------------------------------------------------------------
// pw offsets (ushorts)
#define PW0_OFF  0
#define PW1_OFF  8192
#define PW2_OFF  28672
#define PW3_OFF  32768
#define PWL1_OFF 36864
#define PWL2_OFF 57344

#define SM_CL   0
#define SM_SS   3072
#define SM_SE   3136
#define SM_HA   3200
#define SM_HB   5248
#define SM_S0   7296
#define SM_V1   13440
#define SM_V2   44160
#define SM_SIZE 95360

__global__ __launch_bounds__(128) void edge_out(
    const float* __restrict__ vec, const float* __restrict__ x,
    const float* __restrict__ Vin, const int* __restrict__ senders,
    const unsigned short* __restrict__ pw, const float* __restrict__ gconsts,
    const float* __restrict__ node_sum, float* __restrict__ xout,
    float* __restrict__ vout) {
  extern __shared__ char smem[];
  float* cL = (float*)(smem + SM_CL);
  int* sS = (int*)(smem + SM_SS);
  float* sE = (float*)(smem + SM_SE);
  unsigned short* hA = (unsigned short*)(smem + SM_HA);
  unsigned short* hB = (unsigned short*)(smem + SM_HB);
  unsigned short* s0w = (unsigned short*)(smem + SM_S0);   // 16 x 192
  unsigned short* v1w = (unsigned short*)(smem + SM_V1);   // 3 planes x 16x320
  unsigned short* v2w = (unsigned short*)(smem + SM_V2);   // 5 planes x 16x320

  int tid = threadIdx.x;
  int e0 = blockIdx.x * 16;

  for (int i = tid; i < NCONSTS; i += 128) cL[i] = gconsts[i];
  if (tid < 16) {
    int e = e0 + tid;
    sS[tid] = senders[e];
    float vx = vec[e * 3], vy = vec[e * 3 + 1], vz = vec[e * 3 + 2];
    float d = sqrtf(vx * vx + vy * vy + vz * vz);
    float d2 = d * d, d6 = d2 * d2 * d2, d7 = d6 * d, d8 = d7 * d;
    sE[tid] = (d < 1.f) ? (1.f - 28.f * d6 + 48.f * d7 - 21.f * d8) : 0.f;
  }
  __syncthreads();

  // zero V_out[:, 0:64]
  #pragma unroll
  for (int j = 0; j < 8; ++j) {
    int idx = tid * 8 + j;
    vout[(size_t)(e0 + (idx >> 6)) * 576 + (idx & 63)] = 0.f;
  }

  const float SQ3 = 1.7320508075688772f, SQ5 = 2.2360679774997896f;
  for (int it = 0; it < 8; ++it) {
    int item = it * 128 + tid;           // (e,u): 16 x 64 items
    int e = item >> 6, u = item & 63;
    float wy[16], vc[9], o[7];
    const float* ns = node_sum + ((size_t)sS[e] * 64 + u) * 16;
    #pragma unroll
    for (int i = 0; i < 16; ++i) wy[i] = ns[i] * 0.25f;  // 1/sqrt(avg_neigh)
    const float* vp = Vin + (size_t)(e0 + e) * 576 + u * 9;
    #pragma unroll
    for (int j = 0; j < 9; ++j) vc[j] = vp[j];

    unsigned short* s0r = s0w + e * 192 + u * 3;
    tpath<0, 0, 0>(wy, vc, cL + OFF_P000, 1.f, o); s0r[0] = bfbits(o[0]);
    tpath<1, 1, 0>(wy, vc, cL + OFF_P110, 1.f, o); s0r[1] = bfbits(o[0]);
    tpath<2, 2, 0>(wy, vc, cL + OFF_P220, 1.f, o); s0r[2] = bfbits(o[0]);

    int m5 = e * 320 + u * 5;
    tpath<0, 1, 1>(wy, vc, cL + OFF_P011, SQ3, o);
    for (int i = 0; i < 3; ++i) v1w[i * 5120 + m5 + 0] = bfbits(o[i]);
    tpath<1, 0, 1>(wy, vc, cL + OFF_P101, SQ3, o);
    for (int i = 0; i < 3; ++i) v1w[i * 5120 + m5 + 1] = bfbits(o[i]);
    tpath<1, 2, 1>(wy, vc, cL + OFF_P121, SQ3, o);
    for (int i = 0; i < 3; ++i) v1w[i * 5120 + m5 + 2] = bfbits(o[i]);
    tpath<2, 1, 1>(wy, vc, cL + OFF_P211, SQ3, o);
    for (int i = 0; i < 3; ++i) v1w[i * 5120 + m5 + 3] = bfbits(o[i]);
    tpath<3, 2, 1>(wy, vc, cL + OFF_P321, SQ3, o);
    for (int i = 0; i < 3; ++i) v1w[i * 5120 + m5 + 4] = bfbits(o[i]);

    tpath<0, 2, 2>(wy, vc, cL + OFF_P022, SQ5, o);
    for (int k = 0; k < 5; ++k) v2w[k * 5120 + m5 + 0] = bfbits(o[k]);
    tpath<1, 1, 2>(wy, vc, cL + OFF_P112, SQ5, o);
    for (int k = 0; k < 5; ++k) v2w[k * 5120 + m5 + 1] = bfbits(o[k]);
    tpath<2, 0, 2>(wy, vc, cL + OFF_P202, SQ5, o);
    for (int k = 0; k < 5; ++k) v2w[k * 5120 + m5 + 2] = bfbits(o[k]);
    tpath<2, 2, 2>(wy, vc, cL + OFF_P222, SQ5, o);
    for (int k = 0; k < 5; ++k) v2w[k * 5120 + m5 + 3] = bfbits(o[k]);
    tpath<3, 1, 2>(wy, vc, cL + OFF_P312, SQ5, o);
    for (int k = 0; k < 5; ++k) v2w[k * 5120 + m5 + 4] = bfbits(o[k]);
  }
  __syncthreads();

  int wave = tid >> 5, lane = tid & 31;
  int hi = (lane & 16) ? 8 : 0;
  const float invs320 = 0.05590169943749474f;  // 1/sqrt(320)

  if (wave == 0) {
    // ---- scalar MLP: x2@W1 -> silu_n -> @W2 -> silu_n -> @W3 -> envelope ----
    float C = cL[OFF_SILU];
    #pragma unroll
    for (int nt = 0; nt < 4; ++nt) {
      v8f acc = zero8();
      for (int kc = 0; kc < 10; ++kc) {
        v16bf a = (kc < 4) ? ldA_f32(x + (size_t)e0 * 128, 128, kc * 32, lane)
                           : ldA_bf(s0w, 192, kc * 32 - 128, lane);
        v16bf b = ldB(pw + PW1_OFF, kc * 4 + nt, lane);
        acc = wmma_bf(a, b, acc);
      }
      #pragma unroll
      for (int r = 0; r < 8; ++r) {
        float z = acc[r] * invs320;
        float sv = z / (1.f + expf(-z)) * C;
        hA[(r + hi) * 64 + nt * 16 + (lane & 15)] = bfbits(sv);
      }
    }
    #pragma unroll
    for (int nt = 0; nt < 4; ++nt) {
      v8f acc = zero8();
      for (int kc = 0; kc < 2; ++kc) {
        v16bf a = ldA_bf(hA, 64, kc * 32, lane);
        v16bf b = ldB(pw + PW2_OFF, kc * 4 + nt, lane);
        acc = wmma_bf(a, b, acc);
      }
      #pragma unroll
      for (int r = 0; r < 8; ++r) {
        float z = acc[r] * 0.125f;
        float sv = z / (1.f + expf(-z)) * C;
        hB[(r + hi) * 64 + nt * 16 + (lane & 15)] = bfbits(sv);
      }
    }
    #pragma unroll
    for (int nt = 0; nt < 4; ++nt) {
      v8f acc = zero8();
      for (int kc = 0; kc < 2; ++kc) {
        v16bf a = ldA_bf(hB, 64, kc * 32, lane);
        v16bf b = ldB(pw + PW3_OFF, kc * 4 + nt, lane);
        acc = wmma_bf(a, b, acc);
      }
      #pragma unroll
      for (int r = 0; r < 8; ++r) {
        int m = r + hi;
        xout[(size_t)(e0 + m) * 64 + nt * 16 + (lane & 15)] =
            acc[r] * 0.125f * sE[m];
      }
    }
  } else {
    // ---- equivariant linears: 8 GEMMs (3x o1, 5x o2) x 4 n-tiles ----
    for (int idx = wave - 1; idx < 32; idx += 3) {
      int g = idx >> 2, nt = idx & 3;
      const unsigned short* Ap = (g < 3) ? (v1w + g * 5120) : (v2w + (g - 3) * 5120);
      const unsigned short* pB = (g < 3) ? (pw + PWL1_OFF) : (pw + PWL2_OFF);
      v8f acc = zero8();
      for (int kc = 0; kc < 10; ++kc) {
        v16bf a = ldA_bf(Ap, 320, kc * 32, lane);
        v16bf b = ldB(pB, kc * 4 + nt, lane);
        acc = wmma_bf(a, b, acc);
      }
      #pragma unroll
      for (int r = 0; r < 8; ++r) {
        int m = r + hi;
        int up = nt * 16 + (lane & 15);
        int col = (g < 3) ? (64 + up * 3 + g) : (256 + up * 5 + (g - 3));
        vout[(size_t)(e0 + m) * 576 + col] = acc[r] * invs320;
      }
    }
  }
}

// ---------------------------------------------------------------------------
// Launch
// ---------------------------------------------------------------------------
extern "C" void kernel_launch(void* const* d_in, const int* in_sizes, int n_in,
                              void* d_out, int out_size, void* d_ws, size_t ws_size,
                              hipStream_t stream) {
  const float* vec = (const float*)d_in[0];
  const float* x   = (const float*)d_in[1];
  const float* V   = (const float*)d_in[2];
  const int*   snd = (const int*)d_in[3];
  const float* W0  = (const float*)d_in[4];
  const float* W1  = (const float*)d_in[5];
  const float* W2  = (const float*)d_in[6];
  const float* W3  = (const float*)d_in[7];
  const float* Wl1 = (const float*)d_in[8];
  const float* Wl2 = (const float*)d_in[9];

  const int E = in_sizes[3];          // 32768
  const int tiles = E / 16;           // 2048

  float* consts = (float*)d_ws;
  unsigned short* pw = (unsigned short*)((char*)d_ws + 4096);
  float* node = (float*)((char*)d_ws + 163840);      // 2048*64*16 f32 = 8MB
  float* xout = (float*)d_out;
  float* vout = xout + (size_t)E * 64;

  (void)hipFuncSetAttribute((const void*)edge_out,
                            hipFuncAttributeMaxDynamicSharedMemorySize, SM_SIZE);

  build_consts<<<1, 32, 0, stream>>>(consts);
  pack_b<<<4, 128, 0, stream>>>(W0, pw + PW0_OFF, 128);
  pack_b<<<10, 128, 0, stream>>>(W1, pw + PW1_OFF, 320);
  pack_b<<<2, 128, 0, stream>>>(W2, pw + PW2_OFF, 64);
  pack_b<<<2, 128, 0, stream>>>(W3, pw + PW3_OFF, 64);
  pack_b<<<10, 128, 0, stream>>>(Wl1, pw + PWL1_OFF, 320);
  pack_b<<<10, 128, 0, stream>>>(Wl2, pw + PWL2_OFF, 320);
  zero_f32<<<1024, 256, 0, stream>>>(node, 2048 * 64 * 16);
  edge_scatter<<<tiles / 8, 256, 0, stream>>>(vec, x, snd, pw + PW0_OFF, consts, node);
  edge_out<<<tiles, 128, SM_SIZE, stream>>>(vec, x, V, snd, pw, consts, node,
                                            xout, vout);
}
